// MixedMlp_43026982372010
// MI455X (gfx1250) — compile-verified, matrix operand order checked by reading, not run on hardware
//
#include <hip/hip_runtime.h>
#include <hip/hip_bf16.h>

typedef _Float16 v16h __attribute__((ext_vector_type(16)));
typedef _Float16 v8h  __attribute__((ext_vector_type(8)));
typedef float    v8f  __attribute__((ext_vector_type(8)));

#define B_TOT  8192
#define LATENT 32
#define CIN    128
#define HID    256
#define NACT   16
#define NEXP   8
#define GH     128
#define IN0    160    // LATENT + CIN
#define INTER  288    // HID + LATENT
#define LN_EPS 1e-5f

#define TPB    64     // 2 waves cooperating on one 16-row tile

// ---------------------------------------------------------------------------
// helpers
// ---------------------------------------------------------------------------
__device__ __forceinline__ float elu_f(float v) {
  return v > 0.f ? v : (__expf(v) - 1.f);
}

__device__ __forceinline__ v8f wmma16(v16h a, v16h b, v8f c) {
  // D = A(16x32 f16) * B(32x16 f16) + C(16x16 f32)
  return __builtin_amdgcn_wmma_f32_16x16x32_f16(false, a, false, b, (short)0, c,
                                                false, false);
}

// A fragment for 16x32 f16: lane l (group g = l>>4) holds row M=l&15,
// halves 0..7 -> K = kt*32 + 8g + 0..7 ; halves 8..15 -> K = kt*32 + 8g + 16..23
__device__ __forceinline__ v16h load_a_frag(const _Float16* rowbase, int kt, int g) {
  const _Float16* p = rowbase + kt * 32 + 8 * g;
  v8h lo = *(const v8h*)(p);
  v8h hi = *(const v8h*)(p + 16);
  v16h r;
#pragma unroll
  for (int i = 0; i < 8; ++i) { r[i] = lo[i]; r[i + 8] = hi[i]; }
  return r;
}

// ---------------------------------------------------------------------------
// weight pack: fp32 [E][K][srcN] row-major  ->  f16 WMMA B-fragments
// fragment order: e-major, then nt, then kt; within fragment: lane*16 + h
// element (lane,h) of fragment (kt,nt):  K = kt*32 + h + 8*(g + (h>=8)),
//                                        N = nt*16 + (lane&15)
// ---------------------------------------------------------------------------
__global__ void pack_kernel(const float* __restrict__ src, _Float16* __restrict__ dst,
                            int E_, int K, int N, int srcN) {
  int t = blockIdx.x * blockDim.x + threadIdx.x;
  int total = E_ * K * N;
  if (t >= total) return;
  int h    = t & 15;
  int lane = (t >> 4) & 31;
  int frag = t >> 9;
  int KT = K >> 5, NT = N >> 4;
  int pf = KT * NT;
  int e  = frag / pf;
  int f  = frag - e * pf;
  int nt = f / KT;
  int kt = f - nt * KT;
  int g  = lane >> 4, ln = lane & 15;
  int k  = kt * 32 + h + 8 * (g + (h >= 8 ? 1 : 0));
  int n  = nt * 16 + ln;
  float v = (n < srcN) ? src[((size_t)e * K + k) * srcN + n] : 0.f;
  dst[t] = (_Float16)v;
}

// ---------------------------------------------------------------------------
// gate layer: out = elu(inp @ W + b), f16 out to LDS. nt range per wave.
// Dual accumulator chains so the XDL pipe has two independent streams.
// ---------------------------------------------------------------------------
template <int KDIM, int NDIM>
__device__ __forceinline__ void gate_layer(const _Float16* inp, int in_stride,
                                           const v16h* __restrict__ wf,
                                           const float* __restrict__ bias,
                                           _Float16* out, int out_stride,
                                           int nt0, int nt1, int lane) {
  constexpr int KT = KDIM / 32;
  const int g = lane >> 4, ln = lane & 15;
  v16h a[KT];
#pragma unroll
  for (int kt = 0; kt < KT; ++kt) a[kt] = load_a_frag(inp + ln * in_stride, kt, g);
  for (int nt = nt0; nt < nt1; ++nt) {
    v8f acc0 = {}, acc1 = {};
#pragma unroll
    for (int kt = 0; kt < KT; ++kt) {
      v16h b = wf[(nt * KT + kt) * 32 + lane];
      if (kt & 1) acc1 = wmma16(a[kt], b, acc1);
      else        acc0 = wmma16(a[kt], b, acc0);
    }
    int N = nt * 16 + ln;
    float b = bias[N];
#pragma unroll
    for (int i = 0; i < 8; ++i) {
      int M = i + 8 * g;
      out[M * out_stride + N] = (_Float16)elu_f(acc0[i] + acc1[i] + b);
    }
  }
}

// ---------------------------------------------------------------------------
// expert layer (layers 0/1): out = elu( sum_e coeff_e * (inp @ W_e + b_e) )
// nt range per wave; f16 result written into LDS at column offset out_col0.
// ---------------------------------------------------------------------------
template <int KDIM, int NDIM>
__device__ __forceinline__ void expert_layer(const _Float16* inp, int in_stride,
                                             const v16h* __restrict__ wf,
                                             const float* __restrict__ bias,
                                             const float* coeff,      // [16][NEXP]
                                             _Float16* out_lds, int out_stride,
                                             int out_col0, int nt0, int nt1,
                                             int lane) {
  constexpr int KT = KDIM / 32;
  constexpr int NT = NDIM / 16;
  constexpr int FRAGS = KT * NT;
  const int g = lane >> 4, ln = lane & 15;

  v16h a[KT];
#pragma unroll
  for (int kt = 0; kt < KT; ++kt) a[kt] = load_a_frag(inp + ln * in_stride, kt, g);

  float ce[NEXP][8];
#pragma unroll
  for (int e = 0; e < NEXP; ++e)
#pragma unroll
    for (int i = 0; i < 8; ++i) ce[e][i] = coeff[(i + 8 * g) * NEXP + e];

  for (int nt = nt0; nt < nt1; ++nt) {
    v8f mix = {};
#pragma unroll
    for (int e = 0; e < NEXP; ++e) {
      const v16h* we = wf + (size_t)e * FRAGS * 32;
      v8f acc0 = {}, acc1 = {};
#pragma unroll
      for (int kt = 0; kt < KT; ++kt) {
        v16h b = we[(nt * KT + kt) * 32 + lane];
        if (kt & 1) acc1 = wmma16(a[kt], b, acc1);
        else        acc0 = wmma16(a[kt], b, acc0);
      }
      float be = bias[e * NDIM + nt * 16 + ln];
#pragma unroll
      for (int i = 0; i < 8; ++i) mix[i] += ce[e][i] * (acc0[i] + acc1[i] + be);
    }
#pragma unroll
    for (int i = 0; i < 8; ++i) {
      int M = i + 8 * g, N = nt * 16 + ln;
      out_lds[M * out_stride + out_col0 + N] = (_Float16)elu_f(mix[i]);
    }
  }
}

// ---------------------------------------------------------------------------
// fused MoE kernel: one workgroup == one 16-row batch tile, 2 waves split N
// ---------------------------------------------------------------------------
__global__ __launch_bounds__(TPB) void moe_kernel(
    const float* __restrict__ z, const float* __restrict__ c,
    const float* __restrict__ b0, const float* __restrict__ b1,
    const float* __restrict__ b2, const float* __restrict__ g0b,
    const float* __restrict__ g1b, const float* __restrict__ g2b,
    const float* __restrict__ ln_g, const float* __restrict__ ln_b,
    const v16h* __restrict__ w0f, const v16h* __restrict__ w1f,
    const v16h* __restrict__ w2f, const v16h* __restrict__ g0f,
    const v16h* __restrict__ g1f, const v16h* __restrict__ g2f,
    float* __restrict__ out) {
  __shared__ __attribute__((aligned(16))) _Float16 sx[16][IN0];     // [z|cn]
  __shared__ __attribute__((aligned(16))) _Float16 sg[2][16][GH];   // h0,h1
  __shared__ __attribute__((aligned(16))) _Float16 si0[16][INTER];  // [z|l0]
  __shared__ __attribute__((aligned(16))) _Float16 si1[16][INTER];  // [z|l1]
  __shared__ __attribute__((aligned(16))) float scoef[16][NEXP];
  __shared__ __attribute__((aligned(16))) float spart[2][16][NACT]; // layer2 halves

  const int tid  = threadIdx.x;
  const int w    = tid >> 5;          // wave id (uniform per wave)
  const int lane = tid & 31;
  const int row0 = blockIdx.x * 16;
  const int g = lane >> 4, ln = lane & 15;

  // ---- Phase 1: LayerNorm(c) + stage z/cn as f16 (4 lanes per row) ----
  {
    int r = tid >> 2, q = tid & 3;    // row 0..15, quarter 0..3 (32 cols each)
    int grow = row0 + r;
    const float* crow = c + (size_t)grow * CIN + q * 32;
    float sum = 0.f, sq = 0.f;
#pragma unroll
    for (int j = 0; j < 8; ++j) {
      float4 v4 = ((const float4*)crow)[j];
      sum += v4.x + v4.y + v4.z + v4.w;
      sq  += v4.x * v4.x + v4.y * v4.y + v4.z * v4.z + v4.w * v4.w;
    }
    sum += __shfl_xor(sum, 1); sum += __shfl_xor(sum, 2);
    sq  += __shfl_xor(sq, 1);  sq  += __shfl_xor(sq, 2);
    float mean = sum * (1.f / CIN);
    float var  = sq * (1.f / CIN) - mean * mean;
    float rstd = rsqrtf(var + LN_EPS);
#pragma unroll
    for (int j = 0; j < 8; ++j) {
      float4 v4 = ((const float4*)crow)[j];
      int col = q * 32 + 4 * j;
      sx[r][LATENT + col + 0] = (_Float16)((v4.x - mean) * rstd * ln_g[col + 0] + ln_b[col + 0]);
      sx[r][LATENT + col + 1] = (_Float16)((v4.y - mean) * rstd * ln_g[col + 1] + ln_b[col + 1]);
      sx[r][LATENT + col + 2] = (_Float16)((v4.z - mean) * rstd * ln_g[col + 2] + ln_b[col + 2]);
      sx[r][LATENT + col + 3] = (_Float16)((v4.w - mean) * rstd * ln_g[col + 3] + ln_b[col + 3]);
    }
    const float* zrow = z + (size_t)grow * LATENT;
#pragma unroll
    for (int j = 0; j < 8; ++j) {
      int col = q * 8 + j;
      _Float16 zh = (_Float16)zrow[col];
      sx[r][col]  = zh;
      si0[r][col] = zh;
      si1[r][col] = zh;
    }
  }
  __syncthreads();

  // ---- Phase 2: gate MLP (N-tiles split across the 2 waves) + softmax ----
  gate_layer<IN0, GH>(&sx[0][0], IN0, g0f, g0b, &sg[0][0][0], GH,
                      w * 4, w * 4 + 4, lane);
  __syncthreads();
  gate_layer<GH, GH>(&sg[0][0][0], GH, g1f, g1b, &sg[1][0][0], GH,
                     w * 4, w * 4 + 4, lane);
  __syncthreads();
  if (w == 0) {  // wave-uniform branch: full EXEC inside for WMMA
    v16h a[GH / 32];
#pragma unroll
    for (int kt = 0; kt < GH / 32; ++kt)
      a[kt] = load_a_frag(&sg[1][ln][0], kt, g);
    v8f acc0 = {}, acc1 = {};
#pragma unroll
    for (int kt = 0; kt < GH / 32; ++kt) {
      v16h b = g2f[kt * 32 + lane];
      if (kt & 1) acc1 = wmma16(a[kt], b, acc1);
      else        acc0 = wmma16(a[kt], b, acc0);
    }
    if (ln < NEXP) {
      float bb = g2b[ln];
#pragma unroll
      for (int i = 0; i < 8; ++i)
        scoef[i + 8 * g][ln] = acc0[i] + acc1[i] + bb;
    }
  }
  __syncthreads();
  if (tid < 16) {
    float l_[NEXP], m = -1e30f;
#pragma unroll
    for (int e = 0; e < NEXP; ++e) { l_[e] = scoef[tid][e]; m = fmaxf(m, l_[e]); }
    float s = 0.f;
#pragma unroll
    for (int e = 0; e < NEXP; ++e) { l_[e] = __expf(l_[e] - m); s += l_[e]; }
    float inv = 1.f / s;
#pragma unroll
    for (int e = 0; e < NEXP; ++e) scoef[tid][e] = l_[e] * inv;
  }
  __syncthreads();

  // ---- Phase 3: expert layers (N split across waves) ----
  expert_layer<IN0, HID>(&sx[0][0], IN0, w0f, b0, &scoef[0][0],
                         &si0[0][0], INTER, LATENT, w * 8, w * 8 + 8, lane);
  __syncthreads();
  expert_layer<INTER, HID>(&si0[0][0], INTER, w1f, b1, &scoef[0][0],
                           &si1[0][0], INTER, LATENT, w * 8, w * 8 + 8, lane);
  __syncthreads();

  // ---- Layer 2: single 16-wide N tile; split the 8 experts 4/4 per wave ----
  {
    constexpr int KT = INTER / 32;       // 9
    v16h a[KT];
#pragma unroll
    for (int kt = 0; kt < KT; ++kt) a[kt] = load_a_frag(&si1[ln][0], kt, g);
    float ce[4][8];
#pragma unroll
    for (int e = 0; e < 4; ++e)
#pragma unroll
      for (int i = 0; i < 8; ++i) ce[e][i] = scoef[i + 8 * g][4 * w + e];
    v8f mix = {};
#pragma unroll
    for (int e = 0; e < 4; ++e) {
      int ge = 4 * w + e;
      const v16h* we = w2f + (size_t)ge * KT * 32;   // NT==1
      v8f acc0 = {}, acc1 = {};
#pragma unroll
      for (int kt = 0; kt < KT; ++kt) {
        v16h b = we[kt * 32 + lane];
        if (kt & 1) acc1 = wmma16(a[kt], b, acc1);
        else        acc0 = wmma16(a[kt], b, acc0);
      }
      float be = b2[ge * NACT + ln];
#pragma unroll
      for (int i = 0; i < 8; ++i) mix[i] += ce[e][i] * (acc0[i] + acc1[i] + be);
    }
#pragma unroll
    for (int i = 0; i < 8; ++i) spart[w][i + 8 * g][ln] = mix[i];
  }
  __syncthreads();
#pragma unroll
  for (int idx = 0; idx < 256; idx += TPB) {
    int t = idx + tid;
    int M = t >> 4, N = t & 15;
    out[(size_t)(row0 + M) * NACT + N] = spart[0][M][N] + spart[1][M][N];
  }
}

// ---------------------------------------------------------------------------
// launch
// ---------------------------------------------------------------------------
extern "C" void kernel_launch(void* const* d_in, const int* in_sizes, int n_in,
                              void* d_out, int out_size, void* d_ws, size_t ws_size,
                              hipStream_t stream) {
  const float* z   = (const float*)d_in[0];
  const float* c   = (const float*)d_in[1];
  const float* w0  = (const float*)d_in[2];
  const float* b0  = (const float*)d_in[3];
  const float* w1  = (const float*)d_in[4];
  const float* b1  = (const float*)d_in[5];
  const float* w2  = (const float*)d_in[6];
  const float* b2  = (const float*)d_in[7];
  const float* g0w = (const float*)d_in[8];
  const float* g0b = (const float*)d_in[9];
  const float* g1w = (const float*)d_in[10];
  const float* g1b = (const float*)d_in[11];
  const float* g2w = (const float*)d_in[12];
  const float* g2b = (const float*)d_in[13];
  const float* lng = (const float*)d_in[14];
  const float* lnb = (const float*)d_in[15];
  float* out = (float*)d_out;

  // workspace layout (f16 fragment-packed weights), all sizes multiple of 1KB
  _Float16* ws  = (_Float16*)d_ws;
  size_t n_w0 = (size_t)NEXP * IN0 * HID;    // 327680
  size_t n_w1 = (size_t)NEXP * INTER * HID;  // 589824
  size_t n_w2 = (size_t)NEXP * INTER * NACT; // 36864
  size_t n_g0 = (size_t)IN0 * GH;            // 20480
  size_t n_g1 = (size_t)GH * GH;             // 16384
  size_t n_g2 = (size_t)GH * 16;             // 2048 (padded 8 -> 16 cols)
  _Float16* w0f = ws;
  _Float16* w1f = w0f + n_w0;
  _Float16* w2f = w1f + n_w1;
  _Float16* g0f = w2f + n_w2;
  _Float16* g1f = g0f + n_g0;
  _Float16* g2f = g1f + n_g1;

  const int PT = 256;
  pack_kernel<<<(int)((n_w0 + PT - 1) / PT), PT, 0, stream>>>(w0, w0f, NEXP, IN0, HID, HID);
  pack_kernel<<<(int)((n_w1 + PT - 1) / PT), PT, 0, stream>>>(w1, w1f, NEXP, INTER, HID, HID);
  pack_kernel<<<(int)((n_w2 + PT - 1) / PT), PT, 0, stream>>>(w2, w2f, NEXP, INTER, NACT, NACT);
  pack_kernel<<<(int)((n_g0 + PT - 1) / PT), PT, 0, stream>>>(g0w, g0f, 1, IN0, GH, GH);
  pack_kernel<<<(int)((n_g1 + PT - 1) / PT), PT, 0, stream>>>(g1w, g1f, 1, GH, GH, GH);
  pack_kernel<<<(int)((n_g2 + PT - 1) / PT), PT, 0, stream>>>(g2w, g2f, 1, GH, 16, NEXP);

  int grid = B_TOT / 16;  // 512 workgroups, one 16-row tile each, 2 waves
  moe_kernel<<<grid, TPB, 0, stream>>>(z, c, b0, b1, b2, g0b, g1b, g2b, lng, lnb,
                                       (const v16h*)w0f, (const v16h*)w1f,
                                       (const v16h*)w2f, (const v16h*)g0f,
                                       (const v16h*)g1f, (const v16h*)g2f, out);
}
